// GraphCON_GAT_59184649338989
// MI455X (gfx1250) — compile-verified, matrix operand order checked by reading, not run on hardware
//
#include <hip/hip_runtime.h>

// ---------------- problem constants ----------------
#define NNODES   20000
#define NEDGES   320000
#define ETOT     340000        // edges + self loops
#define NFEAT    128
#define NHID     64
#define HEADS    4
#define HIDALL   256           // HEADS*NHID
#define NCLASS   16
#define NEGSLOPE 0.2f
#define EPSC     1e-5f
#define SP1C     1.3132616875182228f   // softplus(1.0)
#define DTC      1.0f
#define KSC      SP1C
#define TWO_ZO   (2.0f * SP1C * SP1C)  // 2*zeta*omega
#define OM2C     (SP1C * SP1C)         // omega^2

typedef __attribute__((ext_vector_type(16))) _Float16 v16h;
typedef __attribute__((ext_vector_type(8)))  float    v8f;

// ---------------- helpers ----------------
__device__ __forceinline__ float eluf(float v)   { return v > 0.f ? v : (expf(v) - 1.f); }
__device__ __forceinline__ float lrelu(float v)  { return v > 0.f ? v : NEGSLOPE * v; }

// order-preserving float <-> uint encoding for atomic max
__device__ __forceinline__ unsigned f2ord(float f) {
  unsigned u = __float_as_uint(f);
  return (u & 0x80000000u) ? ~u : (u | 0x80000000u);
}
__device__ __forceinline__ float ord2f(unsigned o) {
  return (o & 0x80000000u) ? __uint_as_float(o & 0x7FFFFFFFu) : __uint_as_float(~o);
}

__device__ __forceinline__ void edge_sd(const int* __restrict__ ei, int e, int& s, int& d) {
  if (e < NEDGES) { s = ei[e]; d = ei[NEDGES + e]; }
  else            { s = d = e - NEDGES; }
}

// ================= WMMA fragment swizzle (ISA 16-bit 16x32 layout) =================
// For element with within-chunk index klocal = g*16 + hh*8 + w (g,hh in {0,1}, w in [0,8)):
//   lane = (row_or_col % 16) + 16*hh ,  fragment element index = g*8 + w
// A fragments stored [mtile][kchunk][lane][16], B fragments stored [kchunk][ntile][lane][16].

__device__ __forceinline__ void store_a_swz(_Float16* __restrict__ swz, int KC,
                                            int m, int k, float v) {
  int chunk  = k >> 5;
  int klocal = k & 31;
  int hh   = (klocal >> 3) & 1;
  int eidx = ((klocal >> 4) << 3) + (klocal & 7);
  int lane = (m & 15) + (hh << 4);
  int mtile = m >> 4;
  swz[(((size_t)mtile * KC + chunk) * 32 + lane) * 16 + eidx] = (_Float16)v;
}

// one-time: f32 [M,K] row-major -> f16 A-fragments
__global__ __launch_bounds__(256)
void swizzle_a_kernel(const float* __restrict__ A, _Float16* __restrict__ swz,
                      int M, int K) {
  int idx = (int)(blockIdx.x * blockDim.x + threadIdx.x);
  if (idx >= M * K) return;
  store_a_swz(swz, K >> 5, idx / K, idx % K, A[idx]);
}

// one-time: f32 [K,N] row-major -> f16 B-fragments
__global__ __launch_bounds__(256)
void swizzle_b_kernel(const float* __restrict__ B, _Float16* __restrict__ swz,
                      int K, int N) {
  int idx = (int)(blockIdx.x * blockDim.x + threadIdx.x);
  if (idx >= K * N) return;
  int k = idx / N, n = idx % N;
  int chunk  = k >> 5;
  int klocal = k & 31;
  int hh   = (klocal >> 3) & 1;
  int eidx = ((klocal >> 4) << 3) + (klocal & 7);
  int nt   = n >> 4;
  int lane = (n & 15) + (hh << 4);
  int NT   = N >> 4;
  swz[(((size_t)chunk * NT + nt) * 32 + lane) * 16 + eidx] = (_Float16)B[idx];
}

template <int KC>
__device__ __forceinline__ v16h load_frag_a_swz(const _Float16* __restrict__ swz,
                                                int mtile, int chunk, int lane) {
  return *(const v16h*)(swz + (((size_t)mtile * KC + chunk) * 32 + lane) * 16);
}
template <int NT>
__device__ __forceinline__ v16h load_frag_b_swz(const _Float16* __restrict__ swz,
                                                int chunk, int nt, int lane) {
  return *(const v16h*)(swz + (((size_t)chunk * NT + nt) * 32 + lane) * 16);
}

// ---------------- specialized WMMA GEMM: C[M,N] = act(A[M,K] @ B[K,N] + bias) ----------------
// grid.x = M/16 ; blockDim.x = 32 * (N/16) ; wave w handles column tile w.
// Inner loop per chunk: two 32-byte v16h loads + one v_wmma (no conversions).
template <int K, int N, bool RELU, bool HAS_BIAS, bool DUP, bool XSWZ>
__global__ __launch_bounds__(32 * (N / 16))
void wmma_gemm_kernel(const _Float16* __restrict__ Aswz,
                      const _Float16* __restrict__ Bswz,
                      const float* __restrict__ bias,
                      float* __restrict__ C,
                      float* __restrict__ C2,
                      _Float16* __restrict__ xswz) {  // f16 A-fragment mirror of output
  constexpr int KC = K / 32;
  constexpr int NT = N / 16;
  int lane = (int)(threadIdx.x & 31);
  int wave = (int)(threadIdx.x >> 5);
  int mtile = (int)blockIdx.x;
  int m0 = mtile * 16;

  v8f acc = {};
#pragma unroll
  for (int chunk = 0; chunk < KC; ++chunk) {
    v16h a = load_frag_a_swz<KC>(Aswz, mtile, chunk, lane);
    v16h b = load_frag_b_swz<NT>(Bswz, chunk, wave, lane);
    acc = __builtin_amdgcn_wmma_f32_16x16x32_f16(
        /*neg_a=*/false, a, /*neg_b=*/false, b,
        /*c_mod=*/(short)0, acc, /*reuse_a=*/false, /*reuse_b=*/false);
  }

  // D layout: VGPR r -> row m0 + (lane/16)*8 + r, col n0 + lane%16
  int n  = wave * 16 + (lane & 15);
  int mo = (lane >> 4) * 8;
  float bi = HAS_BIAS ? bias[n] : 0.f;
  float* cp  = C + (size_t)(m0 + mo) * N + n;
  float* cp2 = DUP ? (C2 + (size_t)(m0 + mo) * N + n) : nullptr;
#pragma unroll
  for (int r = 0; r < 8; ++r) {
    float v = acc[r] + bi;
    if (RELU) v = fmaxf(v, 0.f);
    cp[(size_t)r * N] = v;
    if (DUP) cp2[(size_t)r * N] = v;
    if (XSWZ) store_a_swz(xswz, NHID / 32, m0 + mo + r, n, v);  // output feeds next GEMM's A
  }
}

// ---------------- attention scores: a_src/a_dst [N, H] ----------------
__global__ __launch_bounds__(256)
void att_score_kernel(const float* __restrict__ h,
                      const float* __restrict__ att_src,
                      const float* __restrict__ att_dst,
                      float* __restrict__ a_src,
                      float* __restrict__ a_dst) {
  int gid = (int)(blockIdx.x * blockDim.x + threadIdx.x);
  if (gid >= NNODES * HEADS) return;
  int n = gid >> 2, hd = gid & 3;
  const float* hp = h + (size_t)n * HIDALL + hd * NHID;
  const float* as = att_src + hd * NHID;
  const float* ad = att_dst + hd * NHID;
  float s = 0.f, d = 0.f;
#pragma unroll 8
  for (int f = 0; f < NHID; ++f) { float v = hp[f]; s += v * as[f]; d += v * ad[f]; }
  a_src[gid] = s;
  a_dst[gid] = d;
}

// ---------------- per-layer init: zero accumulators ----------------
__global__ __launch_bounds__(256)
void init_layer_kernel(float* __restrict__ out_conv,
                       unsigned* __restrict__ emax_ord,
                       float* __restrict__ denom) {
  int idx = (int)(blockIdx.x * blockDim.x + threadIdx.x);
  if (idx < NNODES * HIDALL) out_conv[idx] = 0.f;
  if (idx < NNODES * HEADS) { emax_ord[idx] = 0u; denom[idx] = 0.f; }
}

// ---------------- edge pass 1: e = leaky_relu(a_src[s]+a_dst[d]); segment max ----------------
__global__ __launch_bounds__(256)
void edge_score_kernel(const int* __restrict__ ei,
                       const float* __restrict__ a_src,
                       const float* __restrict__ a_dst,
                       float* __restrict__ e_buf,
                       unsigned* __restrict__ emax_ord) {
  int gid = (int)(blockIdx.x * blockDim.x + threadIdx.x);
  if (gid >= ETOT * HEADS) return;
  int e = gid >> 2, hd = gid & 3;
  int s, d; edge_sd(ei, e, s, d);
  float v = lrelu(a_src[s * HEADS + hd] + a_dst[d * HEADS + hd]);
  e_buf[gid] = v;
  atomicMax(&emax_ord[d * HEADS + hd], f2ord(v));
}

// ---------------- edge pass 2: ex = exp(e - max); segment sum ----------------
__global__ __launch_bounds__(256)
void edge_exp_kernel(const int* __restrict__ ei,
                     float* __restrict__ e_buf,
                     const unsigned* __restrict__ emax_ord,
                     float* __restrict__ denom) {
  int gid = (int)(blockIdx.x * blockDim.x + threadIdx.x);
  if (gid >= ETOT * HEADS) return;
  int e = gid >> 2, hd = gid & 3;
  int s, d; edge_sd(ei, e, s, d);
  float ex = expf(e_buf[gid] - ord2f(emax_ord[d * HEADS + hd]));
  e_buf[gid] = ex;                       // overwrite with unnormalized weight
  atomicAdd(&denom[d * HEADS + hd], ex);
}

// ---------------- edge pass 3: out[dst] += h[src] * alpha (one wave per edge) ----------------
__global__ __launch_bounds__(256)
void scatter_kernel(const int* __restrict__ ei,
                    const float* __restrict__ h,
                    const float* __restrict__ e_buf,
                    const float* __restrict__ denom,
                    float* __restrict__ out_conv) {
  int lane = (int)(threadIdx.x & 31);
  int wave = (int)(threadIdx.x >> 5);
  int e = (int)blockIdx.x * 8 + wave;    // 42500 blocks * 8 waves == ETOT exactly
  int s, d; edge_sd(ei, e, s, d);
  const float4* h4 = (const float4*)(h + (size_t)s * HIDALL);
  float4*       o4 = (float4*)(out_conv + (size_t)d * HIDALL);
  const float*  eb = e_buf + (size_t)e * HEADS;
  const float*  dn = denom + (size_t)d * HEADS;
  __builtin_prefetch(h4 + lane, 0, 3);   // global_prefetch_b8
#pragma unroll
  for (int it = 0; it < 2; ++it) {
    int q = lane + it * 32;              // float4 index 0..63 ; head = q/16
    int hd = q >> 4;
    float alpha = eb[hd] / (dn[hd] + 1e-16f);
    float4 v = h4[q];
    float* op = (float*)(o4 + q);
    atomicAdd(op + 0, v.x * alpha);
    atomicAdd(op + 1, v.y * alpha);
    atomicAdd(op + 2, v.z * alpha);
    atomicAdd(op + 3, v.w * alpha);
  }
}

// ---------------- node update: elu + head-mean + GraphCON ODE + RMS norms ----------------
// one wave per node; lane handles features f = lane and lane+32.
// Also refreshes the f16 A-fragment mirror of X for the next GEMM.
__global__ __launch_bounds__(256)
void node_update_kernel(const float* __restrict__ out_conv,
                        const float* __restrict__ b_gat,
                        float* __restrict__ X,
                        float* __restrict__ Y,
                        _Float16* __restrict__ xswz) {
  int lane = (int)(threadIdx.x & 31);
  int wave = (int)(threadIdx.x >> 5);
  int n = (int)blockIdx.x * 8 + wave;    // 2500 blocks * 8 waves == 20000 exactly

  const float4* c4 = (const float4*)(out_conv + (size_t)n * HIDALL);
  const float4* b4 = (const float4*)b_gat;

  float xn[2], yn[2];
  float xs = 0.f, ys = 0.f;
#pragma unroll
  for (int i = 0; i < 2; ++i) {
    int f = lane + i * 32;
    // agg[f] = mean over j of elu(conv[n, 4f+j])   (reference's head/feature remix)
    float4 c = c4[f], b = b4[f];
    float agg = 0.25f * (eluf(c.x + b.x) + eluf(c.y + b.y) + eluf(c.z + b.z) + eluf(c.w + b.w));
    float y = Y[(size_t)n * NHID + f];
    float x = X[(size_t)n * NHID + f];
    float ynew = y + DTC * (KSC * agg - TWO_ZO * y - OM2C * x);
    float xnew = x + DTC * ynew;
    xn[i] = xnew; yn[i] = ynew;
    xs += xnew * xnew; ys += ynew * ynew;
  }
  // wave32 reduction
#pragma unroll
  for (int off = 16; off >= 1; off >>= 1) {
    xs += __shfl_xor(xs, off, 32);
    ys += __shfl_xor(ys, off, 32);
  }
  float xinv = 1.f / sqrtf(xs * (1.f / NHID) + EPSC);
  float yinv = 1.f / sqrtf(ys * (1.f / NHID) + EPSC);
#pragma unroll
  for (int i = 0; i < 2; ++i) {
    int f = lane + i * 32;
    float xv = xn[i] * xinv;
    X[(size_t)n * NHID + f] = xv;
    Y[(size_t)n * NHID + f] = yn[i] * yinv;
    store_a_swz(xswz, NHID / 32, n, f, xv);
  }
}

// ---------------- host launcher ----------------
extern "C" void kernel_launch(void* const* d_in, const int* in_sizes, int n_in,
                              void* d_out, int out_size, void* d_ws, size_t ws_size,
                              hipStream_t stream) {
  (void)in_sizes; (void)n_in; (void)out_size; (void)ws_size;

  const float* x        = (const float*)d_in[0];
  const int*   ei       = (const int*)d_in[1];   // [2, NEDGES] flat (src row, then dst row)
  const float* W_enc    = (const float*)d_in[2];
  const float* b_enc    = (const float*)d_in[3];
  const float* W_gat    = (const float*)d_in[4];
  const float* att_src  = (const float*)d_in[5];
  const float* att_dst  = (const float*)d_in[6];
  const float* b_gat    = (const float*)d_in[7];
  const float* W_dec    = (const float*)d_in[8];
  const float* b_dec    = (const float*)d_in[9];
  float* out = (float*)d_out;

  // carve workspace (all layer-resident buffers fit comfortably in 192 MB L2)
  char* ws = (char*)d_ws;
  size_t off = 0;
  auto carve = [&](size_t bytes) -> void* {
    void* p = ws + off;
    off = (off + bytes + 255) & ~(size_t)255;
    return p;
  };
  float*    X        = (float*)carve((size_t)NNODES * NHID * 4);     // 5.12 MB
  float*    Y        = (float*)carve((size_t)NNODES * NHID * 4);     // 5.12 MB
  float*    h        = (float*)carve((size_t)NNODES * HIDALL * 4);   // 20.5 MB
  float*    out_conv = (float*)carve((size_t)NNODES * HIDALL * 4);   // 20.5 MB
  float*    a_src    = (float*)carve((size_t)NNODES * HEADS * 4);
  float*    a_dst    = (float*)carve((size_t)NNODES * HEADS * 4);
  float*    e_buf    = (float*)carve((size_t)ETOT * HEADS * 4);      // 5.44 MB
  unsigned* emax     = (unsigned*)carve((size_t)NNODES * HEADS * 4);
  float*    denom    = (float*)carve((size_t)NNODES * HEADS * 4);
  _Float16* x_swz    = (_Float16*)carve((size_t)NNODES * NFEAT * 2); // 5.12 MB
  _Float16* X_swz    = (_Float16*)carve((size_t)NNODES * NHID * 2);  // 2.56 MB
  _Float16* wenc_swz = (_Float16*)carve((size_t)NFEAT * NHID * 2);   // 16 KB
  _Float16* wgat_swz = (_Float16*)carve((size_t)NHID * HIDALL * 2);  // 32 KB
  _Float16* wdec_swz = (_Float16*)carve((size_t)NHID * NCLASS * 2);  // 2 KB

  // one-time pre-swizzle into WMMA fragment order (f16)
  swizzle_a_kernel<<<(NNODES * NFEAT + 255) / 256, 256, 0, stream>>>(x, x_swz, NNODES, NFEAT);
  swizzle_b_kernel<<<(NFEAT * NHID + 255) / 256, 256, 0, stream>>>(W_enc, wenc_swz, NFEAT, NHID);
  swizzle_b_kernel<<<(NHID * HIDALL + 255) / 256, 256, 0, stream>>>(W_gat, wgat_swz, NHID, HIDALL);
  swizzle_b_kernel<<<(NHID * NCLASS + 255) / 256, 256, 0, stream>>>(W_dec, wdec_swz, NHID, NCLASS);

  const int MT = NNODES / 16;  // 1250 row tiles, exact

  // encoder: Y = relu(x @ W_enc + b_enc); X = Y (f32 + f16-swizzled mirror)
  wmma_gemm_kernel<NFEAT, NHID, true, true, true, true>
      <<<MT, 128, 0, stream>>>(x_swz, wenc_swz, b_enc, Y, X, X_swz);

  for (int layer = 0; layer < 3; ++layer) {
    // h = X @ W_gat  (K=64 -> 2 wmma; N=256 -> 16 waves/block)
    wmma_gemm_kernel<NHID, HIDALL, false, false, false, false>
        <<<MT, 512, 0, stream>>>(X_swz, wgat_swz, nullptr, h, nullptr, nullptr);
    att_score_kernel<<<(NNODES * HEADS + 255) / 256, 256, 0, stream>>>(h, att_src, att_dst,
                                                                       a_src, a_dst);
    init_layer_kernel<<<(NNODES * HIDALL) / 256, 256, 0, stream>>>(out_conv, emax, denom);
    edge_score_kernel<<<(ETOT * HEADS + 255) / 256, 256, 0, stream>>>(ei, a_src, a_dst,
                                                                      e_buf, emax);
    edge_exp_kernel<<<(ETOT * HEADS + 255) / 256, 256, 0, stream>>>(ei, e_buf, emax, denom);
    scatter_kernel<<<ETOT / 8, 256, 0, stream>>>(ei, h, e_buf, denom, out_conv);
    node_update_kernel<<<NNODES / 8, 256, 0, stream>>>(out_conv, b_gat, X, Y, X_swz);
  }

  // decoder: out = X @ W_dec + b_dec   (K=64 -> 2 wmma; N=16 -> 1 wave/block)
  wmma_gemm_kernel<NHID, NCLASS, false, true, false, false>
      <<<MT, 32, 0, stream>>>(X_swz, wdec_swz, b_dec, out, nullptr, nullptr);
}
